// DecoderFactoredLSTM_27547920236684
// MI455X (gfx1250) — matrix operand order repaired and checked by reading, not compiled, not run on hardware
//
#include <hip/hip_runtime.h>
#include <hip/hip_bf16.h>
#include <math.h>

// ---------------- CDNA5 WMMA types ----------------
typedef _Float16 v16h __attribute__((ext_vector_type(16)));
typedef float    v8f  __attribute__((ext_vector_type(8)));

union F16Frag { v16h v; uint4 q[2]; };

#define BATCH   64
#define TSTEPS  40
#define EMBED   512
#define HID     1024
#define FACT    512
#define VOCAB   32000
#define ROWS    (BATCH * TSTEPS)   // 2560
#define GSZ     (BATCH * HID)      // 65536
#define KSPLIT  4                  // split-K factor for the recurrence GEMM

// ---------------------------------------------------------------------------
// Convert weight matrix W[N][K] (f32, row-major, out = x·W^T) into f16
// WMMA-B-fragment-tiled layout:
//   Bt[ ((n/16)*(K/32) + k/32)*512 + ((n&15) + 16*((k&31)>>4))*16 + (k&15) ]
// so a GEMM lane loads its 16 B-fragment halfwords as 32 contiguous bytes.
// ---------------------------------------------------------------------------
__global__ __launch_bounds__(256) void convert_b_tiled(
    const float* __restrict__ W, long long sWz,
    _Float16* __restrict__ Bt, long long sBz,
    int N, int K)
{
    const int z = blockIdx.z;
    W  += (size_t)z * sWz;
    Bt += (size_t)z * sBz;
    size_t gid = (size_t)blockIdx.x * 256 + threadIdx.x;
    if (gid >= (size_t)N * K) return;
    int n = (int)(gid / K);
    int k = (int)(gid % K);
    size_t tile = (size_t)(n >> 4) * (K >> 5) + (k >> 5);
    int lane = (n & 15) + (((k & 31) >> 4) << 4);
    int j = k & 15;
    Bt[tile * 512 + lane * 16 + j] = (_Float16)W[(size_t)n * K + k];
}

// ---------------------------------------------------------------------------
// Embedding gather + f32 -> f16, output row-major [t*64+b][512]
// ---------------------------------------------------------------------------
__global__ __launch_bounds__(128) void embed_gather(
    const int* __restrict__ cap, const float* __restrict__ Bw,
    _Float16* __restrict__ X)
{
    int row = blockIdx.x;             // t*64 + b
    int t = row >> 6, b = row & 63;
    int idx = cap[b * TSTEPS + t];
    int e = threadIdx.x;              // 0..127, 4 elems each
    float4 f = ((const float4*)(Bw + (size_t)idx * EMBED))[e];
    _Float16* dst = X + (size_t)row * EMBED + e * 4;
    dst[0] = (_Float16)f.x; dst[1] = (_Float16)f.y;
    dst[2] = (_Float16)f.z; dst[3] = (_Float16)f.w;
}

// ---------------------------------------------------------------------------
// Batched f16 WMMA GEMM, f32 accumulate.
//   out[m][n] = sum_k A[m][k] * W[n][k]  (+ bias[n]) (+ add32[m][n])
// A: row-major f16 [M][K];  Bt: fragment-tiled f16.
// Block = 256 threads = 8 waves -> 64(M) x 256(N); wave tile 32x64
// (8 WMMA : 12 b128 loads per k-step).
// ---------------------------------------------------------------------------
__global__ __launch_bounds__(256) void gemm_wmma_f16(
    const _Float16* __restrict__ A, long long sAz, int K,
    const _Float16* __restrict__ Bt, long long sBz,
    const float* __restrict__ bias, long long sBiasz,
    const float* __restrict__ add32, long long sAddz,
    float* __restrict__ out32, long long sO32z,
    _Float16* __restrict__ out16, long long sO16z,
    int M, int N)
{
    const int z = blockIdx.z;
    A  += (size_t)z * sAz;
    Bt += (size_t)z * sBz;
    if (bias)  bias  += (size_t)z * sBiasz;
    if (add32) add32 += (size_t)z * sAddz;
    if (out32) out32 += (size_t)z * sO32z;
    if (out16) out16 += (size_t)z * sO16z;

    const int tid  = threadIdx.x;
    const int lane = tid & 31;
    const int wave = tid >> 5;
    const int lr = lane & 15;
    const int lh = lane >> 4;
    const int m_base = blockIdx.y * 64  + (wave >> 2) * 32;
    const int n_base = blockIdx.x * 256 + (wave & 3) * 64;

    v8f acc[2][4] = {};
    const int ktiles = K >> 5;

    for (int kt = 0; kt < ktiles; ++kt) {
        const int k0 = kt << 5;
        F16Frag a[2], b[4];
#pragma unroll
        for (int mt = 0; mt < 2; ++mt) {
            const _Float16* ap = A + (size_t)(m_base + mt * 16 + lr) * K + k0 + lh * 16;
            a[mt].q[0] = *(const uint4*)(ap);
            a[mt].q[1] = *(const uint4*)(ap + 8);
        }
#pragma unroll
        for (int nt = 0; nt < 4; ++nt) {
            const _Float16* bp = Bt +
                ((size_t)((n_base >> 4) + nt) * ktiles + kt) * 512 + lane * 16;
            b[nt].q[0] = *(const uint4*)(bp);
            b[nt].q[1] = *(const uint4*)(bp + 8);
        }
#pragma unroll
        for (int mt = 0; mt < 2; ++mt)
#pragma unroll
            for (int nt = 0; nt < 4; ++nt)
                acc[mt][nt] = __builtin_amdgcn_wmma_f32_16x16x32_f16(
                    false, a[mt].v, false, b[nt].v,
                    (short)0, acc[mt][nt], false, false);
    }

#pragma unroll
    for (int mt = 0; mt < 2; ++mt) {
#pragma unroll
        for (int nt = 0; nt < 4; ++nt) {
            const int n  = n_base + nt * 16 + lr;
            const float bs = bias ? bias[n] : 0.0f;
            const int m0 = m_base + mt * 16 + lh * 8;
#pragma unroll
            for (int r = 0; r < 8; ++r) {
                const size_t off = (size_t)(m0 + r) * N + n;
                float val = acc[mt][nt][r] + bs;
                if (add32) val += add32[off];
                if (out32) out32[off] = val;
                if (out16) out16[off] = (_Float16)val;
            }
        }
    }
}

// ---------------------------------------------------------------------------
// Recurrence step GEMM, split-K specialized:
//   part[s][g][64][1024] = h(64x1024,f16) · W[g]^T  over K-chunk s
// grid = (HID/256, KSPLIT, 4 gates); block = 256 = 8 waves, tile 64x256.
// ---------------------------------------------------------------------------
__global__ __launch_bounds__(256) void lstm_step_gemm(
    const _Float16* __restrict__ h16,      // [64][1024]
    const _Float16* __restrict__ Wt,       // [4] fragment-tiled, K=1024
    float* __restrict__ part)              // [KSPLIT][4][64][1024]
{
    const int g = blockIdx.z;
    const int s = blockIdx.y;
    const _Float16* Bt = Wt + (size_t)g * HID * HID;

    const int tid  = threadIdx.x;
    const int lane = tid & 31;
    const int wave = tid >> 5;
    const int lr = lane & 15;
    const int lh = lane >> 4;
    const int m_base = (wave >> 2) * 32;                   // M = 64, one block-row
    const int n_base = blockIdx.x * 256 + (wave & 3) * 64;

    v8f acc[2][4] = {};
    const int ktiles = HID >> 5;                           // 32 tiles of K=32
    const int kt0 = s * (ktiles / KSPLIT);                 // 8 per split
    const int kt1 = kt0 + (ktiles / KSPLIT);

    for (int kt = kt0; kt < kt1; ++kt) {
        const int k0 = kt << 5;
        F16Frag a[2], b[4];
#pragma unroll
        for (int mt = 0; mt < 2; ++mt) {
            const _Float16* ap = h16 + (size_t)(m_base + mt * 16 + lr) * HID + k0 + lh * 16;
            a[mt].q[0] = *(const uint4*)(ap);
            a[mt].q[1] = *(const uint4*)(ap + 8);
        }
#pragma unroll
        for (int nt = 0; nt < 4; ++nt) {
            const _Float16* bp = Bt +
                ((size_t)((n_base >> 4) + nt) * ktiles + kt) * 512 + lane * 16;
            b[nt].q[0] = *(const uint4*)(bp);
            b[nt].q[1] = *(const uint4*)(bp + 8);
        }
#pragma unroll
        for (int mt = 0; mt < 2; ++mt)
#pragma unroll
            for (int nt = 0; nt < 4; ++nt)
                acc[mt][nt] = __builtin_amdgcn_wmma_f32_16x16x32_f16(
                    false, a[mt].v, false, b[nt].v,
                    (short)0, acc[mt][nt], false, false);
    }

    float* outp = part + ((size_t)s * 4 + g) * GSZ;
#pragma unroll
    for (int mt = 0; mt < 2; ++mt) {
#pragma unroll
        for (int nt = 0; nt < 4; ++nt) {
            const int n  = n_base + nt * 16 + lr;
            const int m0 = m_base + mt * 16 + lh * 8;
#pragma unroll
            for (int r = 0; r < 8; ++r)
                outp[(size_t)(m0 + r) * HID + n] = acc[mt][nt][r];
        }
    }
}

// ---------------------------------------------------------------------------
// Zero h (f16) and c (f32)
// ---------------------------------------------------------------------------
__global__ __launch_bounds__(256) void init_state(_Float16* __restrict__ h,
                                                  float* __restrict__ c)
{
    int i = blockIdx.x * 256 + threadIdx.x;
    if (i < GSZ) { h[i] = (_Float16)0.0f; c[i] = 0.0f; }
}

// ---------------------------------------------------------------------------
// LSTM pointwise: reduce split-K partials, add u_t + W_b, apply gates.
// ---------------------------------------------------------------------------
__global__ __launch_bounds__(256) void lstm_pointwise(
    const float* __restrict__ part,        // [KSPLIT][4][64][1024]
    const float* __restrict__ u32,         // [4][2560][1024]
    const float* __restrict__ W_b,         // [4][1024]
    float* __restrict__ c,
    _Float16* __restrict__ h16,
    _Float16* __restrict__ hiddens, int t)
{
    int i = blockIdx.x * 256 + threadIdx.x;     // 0 .. GSZ-1 : b*1024 + n
    if (i >= GSZ) return;
    const int n = i & (HID - 1);

    float gv[4];
#pragma unroll
    for (int g = 0; g < 4; ++g) {
        float v = W_b[g * HID + n] + u32[(size_t)g * ROWS * HID + (size_t)t * GSZ + i];
#pragma unroll
        for (int s = 0; s < KSPLIT; ++s)
            v += part[((size_t)s * 4 + g) * GSZ + i];
        gv[g] = v;
    }
    float it = 1.0f / (1.0f + expf(-gv[0]));
    float ft = 1.0f / (1.0f + expf(-gv[1]));
    float ot = 1.0f / (1.0f + expf(-gv[2]));
    float ct = tanhf(gv[3]);
    float cn = ft * c[i] + it * ct;
    c[i] = cn;
    float hn = ot * cn;                         // no tanh on c (matches reference)
    h16[i] = (_Float16)hn;
    hiddens[(size_t)t * GSZ + i] = (_Float16)hn;
}

// ---------------------------------------------------------------------------
extern "C" void kernel_launch(void* const* d_in, const int* in_sizes, int n_in,
                              void* d_out, int out_size, void* d_ws, size_t ws_size,
                              hipStream_t stream)
{
    const int*   captions = (const int*)  d_in[0];
    // d_in[1] = lengths (all == T, unused by reference math)
    const float* B_w = (const float*)d_in[2];
    const float* V_w = (const float*)d_in[3];
    const float* V_b = (const float*)d_in[4];
    const float* S_w = (const float*)d_in[5];
    const float* S_b = (const float*)d_in[6];
    const float* U_w = (const float*)d_in[7];
    const float* U_b = (const float*)d_in[8];
    const float* W_w = (const float*)d_in[9];
    const float* W_b = (const float*)d_in[10];
    const float* C_w = (const float*)d_in[11];
    const float* C_b = (const float*)d_in[12];
    float* out = (float*)d_out;

    // -------- workspace carve (all fully written before read) --------
    char* p = (char*)d_ws;
    auto carve = [&](size_t bytes) -> char* {
        char* r = p; p += (bytes + 255) & ~(size_t)255; return r;
    };
    _Float16* Vt   = (_Float16*)carve((size_t)4 * FACT * EMBED * 2);
    _Float16* St   = (_Float16*)carve((size_t)4 * FACT * FACT  * 2);
    _Float16* Ut   = (_Float16*)carve((size_t)4 * HID  * FACT  * 2);
    _Float16* Wt   = (_Float16*)carve((size_t)4 * HID  * HID   * 2);
    _Float16* Ct   = (_Float16*)carve((size_t)VOCAB * HID * 2);
    _Float16* X0   = (_Float16*)carve((size_t)ROWS * EMBED * 2);
    _Float16* v16  = (_Float16*)carve((size_t)4 * ROWS * FACT * 2);
    _Float16* s16  = (_Float16*)carve((size_t)4 * ROWS * FACT * 2);
    float*    u32  = (float*)   carve((size_t)4 * ROWS * HID * 4);
    float*    part = (float*)   carve((size_t)KSPLIT * 4 * GSZ * 4);
    float*    cst  = (float*)   carve((size_t)GSZ * 4);
    _Float16* h16  = (_Float16*)carve((size_t)GSZ * 2);
    _Float16* hid  = (_Float16*)carve((size_t)ROWS * HID * 2);

    // -------- 1. weight conversions to WMMA fragment tiles --------
    convert_b_tiled<<<dim3((FACT*EMBED)/256, 1, 4), 256, 0, stream>>>(
        V_w, (long long)FACT*EMBED, Vt, (long long)FACT*EMBED, FACT, EMBED);
    convert_b_tiled<<<dim3((FACT*FACT)/256, 1, 4), 256, 0, stream>>>(
        S_w, (long long)FACT*FACT, St, (long long)FACT*FACT, FACT, FACT);
    convert_b_tiled<<<dim3((HID*FACT)/256, 1, 4), 256, 0, stream>>>(
        U_w, (long long)HID*FACT, Ut, (long long)HID*FACT, HID, FACT);
    convert_b_tiled<<<dim3((HID*HID)/256, 1, 4), 256, 0, stream>>>(
        W_w, (long long)HID*HID, Wt, (long long)HID*HID, HID, HID);
    convert_b_tiled<<<dim3(((size_t)VOCAB*HID)/256, 1, 1), 256, 0, stream>>>(
        C_w, 0, Ct, 0, VOCAB, HID);

    // -------- 2. embedding gather --------
    embed_gather<<<ROWS, 128, 0, stream>>>(captions, B_w, X0);

    // -------- 3. hoisted input-path GEMMs (gate batched in grid.z) --------
    gemm_wmma_f16<<<dim3(FACT/256, ROWS/64, 4), 256, 0, stream>>>(
        X0, 0, EMBED, Vt, (long long)FACT*EMBED,
        V_b, FACT, nullptr, 0, nullptr, 0,
        v16, (long long)ROWS*FACT, ROWS, FACT);
    gemm_wmma_f16<<<dim3(FACT/256, ROWS/64, 4), 256, 0, stream>>>(
        v16, (long long)ROWS*FACT, FACT, St, (long long)FACT*FACT,
        S_b, FACT, nullptr, 0, nullptr, 0,
        s16, (long long)ROWS*FACT, ROWS, FACT);
    gemm_wmma_f16<<<dim3(HID/256, ROWS/64, 4), 256, 0, stream>>>(
        s16, (long long)ROWS*FACT, FACT, Ut, (long long)HID*FACT,
        U_b, HID, nullptr, 0,
        u32, (long long)ROWS*HID, nullptr, 0, ROWS, HID);

    // -------- 4. recurrence (split-K step GEMM + fused reduce/activate) ----
    init_state<<<GSZ/256, 256, 0, stream>>>(h16, cst);
    for (int t = 0; t < TSTEPS; ++t) {
        lstm_step_gemm<<<dim3(HID/256, KSPLIT, 4), 256, 0, stream>>>(h16, Wt, part);
        lstm_pointwise<<<GSZ/256, 256, 0, stream>>>(part, u32, W_b, cst, h16, hid, t);
    }

    // -------- 5. output projection: logits = hiddens · C^T + C_b --------
    gemm_wmma_f16<<<dim3(VOCAB/256, ROWS/64, 1), 256, 0, stream>>>(
        hid, 0, HID, Ct, 0,
        C_b, 0, nullptr, 0,
        out, 0, nullptr, 0, ROWS, VOCAB);

    (void)in_sizes; (void)n_in; (void)out_size; (void)ws_size;
}
